// DragMeshNetwork_6098853560486
// MI455X (gfx1250) — matrix operand order repaired and checked by reading, not compiled, not run on hardware
//
#include <hip/hip_runtime.h>
#include <math.h>

typedef __attribute__((ext_vector_type(16))) __bf16 v16bf;
typedef __attribute__((ext_vector_type(8)))  float  v8f;
typedef __attribute__((ext_vector_type(4)))  float  v4f;

#if defined(__has_builtin)
#if __has_builtin(__builtin_amdgcn_sched_barrier)
#define SCHED_FENCE() __builtin_amdgcn_sched_barrier(0)
#endif
#endif
#ifndef SCHED_FENCE
#define SCHED_FENCE()
#endif

// ---------------- shared host/device structs ----------------
struct TPHdr {
    int   nT1;          // number of t-program entries for TP1
    float c000a;        // cg(0,0,0) * alpha   (TP2 path 1)
    float c110a[9];     // cg(1,1,0)[i][j] * alpha (TP2 path 2)
    int   pad[6];
};
// t-program entry (int4): {aIdx, bIdx, tSlot, f32bits(coef)}  -- sorted by tSlot
// w-program entry (int4): {outCol, tBase, d3, f32bits(alpha)}

#define TP1_NW   272
#define TP2_NW   32
#define DIM_X1   160
#define DIM_XG   128

__device__ __forceinline__ void atomAddF(float* p, float v) {
    __hip_atomic_fetch_add(p, v, __ATOMIC_RELAXED, __HIP_MEMORY_SCOPE_AGENT);
}

// =====================================================================
// K0z : zero accumulators, transpose+scale weights to bf16 (N-major)
// =====================================================================
__global__ void k0_zero(float* __restrict__ x0, float* __restrict__ x1,
                        float* __restrict__ out,
                        __bf16* __restrict__ W2A, __bf16* __restrict__ W2B,
                        const float* __restrict__ wfc12,   // 256 x 272
                        const float* __restrict__ wfc22,   // 256 x 32
                        int N, int G) {
    int i = blockIdx.x * blockDim.x + threadIdx.x;
    if (i < N * DIM_X1) x1[i] = 0.0f;
    if (i < N * 16)     x0[i] = 0.0f;
    if (i < G)          out[i] = 0.0f;
    if (i < TP1_NW * 256) {              // i = n*256 + k
        int n = i >> 8, k = i & 255;
        W2A[i] = (__bf16)(wfc12[k * TP1_NW + n] * 0.0625f);   // fold 1/sqrt(256)
    }
    if (i < TP2_NW * 256) {
        int n = i >> 8, k = i & 255;
        W2B[i] = (__bf16)(wfc22[k * TP2_NW + n] * 0.0625f);
    }
}

// =====================================================================
// K0p : build Clebsch-Gordan programs on device (1 thread, doubles)
// =====================================================================
__device__ double factd(int n) { double r = 1.0; for (int i = 2; i <= n; ++i) r *= (double)i; return r; }

__device__ double cg_cplx(int j1, int m1, int j2, int m2, int j3, int m3) {
    if (m1 + m2 != m3) return 0.0;
    double pref = sqrt((double)(2 * j3 + 1) * factd(j3 + j1 - j2) * factd(j3 - j1 + j2) *
                       factd(j1 + j2 - j3) / factd(j1 + j2 + j3 + 1));
    pref *= sqrt(factd(j3 + m3) * factd(j3 - m3) * factd(j1 - m1) * factd(j1 + m1) *
                 factd(j2 - m2) * factd(j2 + m2));
    double s = 0.0;
    for (int k = 0; k <= j1 + j2 - j3; ++k) {
        int e0 = k, e1 = j1 + j2 - j3 - k, e2 = j1 - m1 - k;
        int e3 = j2 + m2 - k, e4 = j3 - j2 + m1 + k, e5 = j3 - j1 - m2 + k;
        if (e0 < 0 || e1 < 0 || e2 < 0 || e3 < 0 || e4 < 0 || e5 < 0) continue;
        double den = factd(e0) * factd(e1) * factd(e2) * factd(e3) * factd(e4) * factd(e5);
        s += ((k & 1) ? -1.0 : 1.0) / den;
    }
    return pref * s;
}

struct URow { int n; int col[2]; double re[2], im[2]; };

__device__ void build_U(int l, URow U[7]) {
    int dim = 2 * l + 1;
    for (int i = 0; i < dim; i++) U[i].n = 0;
    U[l].col[0] = l; U[l].re[0] = 1.0; U[l].im[0] = 0.0; U[l].n = 1;
    double r2 = 1.0 / sqrt(2.0);
    for (int m = 1; m <= l; m++) {
        double sg = (m & 1) ? -1.0 : 1.0;
        U[l + m].col[0] = l + m; U[l + m].re[0] = sg * r2; U[l + m].im[0] = 0.0;
        U[l + m].col[1] = l - m; U[l + m].re[1] = r2;      U[l + m].im[1] = 0.0;
        U[l + m].n = 2;
        U[l - m].col[0] = l + m; U[l - m].re[0] = 0.0; U[l - m].im[0] = -sg * r2;
        U[l - m].col[1] = l - m; U[l - m].re[1] = 0.0; U[l - m].im[1] = r2;
        U[l - m].n = 2;
    }
}

__device__ void real_cg_dev(int l1, int l2, int l3, float* T) {
    int d1 = 2 * l1 + 1, d2 = 2 * l2 + 1, d3 = 2 * l3 + 1;
    URow U1[7], U2[7], U3[7];
    build_U(l1, U1); build_U(l2, U2); build_U(l3, U3);
    double Tr[343], Ti[343];
    double sr = 0.0, si = 0.0;
    for (int i = 0; i < d1; i++)
        for (int j = 0; j < d2; j++)
            for (int k = 0; k < d3; k++) {
                double re = 0.0, im = 0.0;
                for (int ai = 0; ai < U1[i].n; ai++)
                    for (int bi = 0; bi < U2[j].n; bi++)
                        for (int ci = 0; ci < U3[k].n; ci++) {
                            int a = U1[i].col[ai], b = U2[j].col[bi], c = U3[k].col[ci];
                            double C = cg_cplx(l1, a - l1, l2, b - l2, l3, c - l3);
                            if (C == 0.0) continue;
                            double u1r = U1[i].re[ai], u1i = U1[i].im[ai];
                            double u2r = U2[j].re[bi], u2i = U2[j].im[bi];
                            double u3r = U3[k].re[ci], u3i = -U3[k].im[ci]; // conj
                            double pr = u1r * u2r - u1i * u2i, pi = u1r * u2i + u1i * u2r;
                            double qr = pr * u3r - pi * u3i, qi = pr * u3i + pi * u3r;
                            re += qr * C; im += qi * C;
                        }
                int idx = (i * d2 + j) * d3 + k;
                Tr[idx] = re; Ti[idx] = im;
                sr += fabs(re); si += fabs(im);
            }
    bool useR = (sr >= si);
    int tot = d1 * d2 * d3;
    double nrm = 0.0;
    for (int idx = 0; idx < tot; idx++) { double v = useR ? Tr[idx] : Ti[idx]; nrm += v * v; }
    nrm = sqrt(nrm);
    double inv = (nrm > 0.0) ? 1.0 / nrm : 0.0;
    for (int idx = 0; idx < tot; idx++) T[idx] = (float)((useR ? Tr[idx] : Ti[idx]) * inv);
}

__global__ void k0_prog(TPHdr* __restrict__ hdr, int4* __restrict__ tprog,
                        int4* __restrict__ wprog) {
    if (threadIdx.x != 0 || blockIdx.x != 0) return;
    const int gm[8] = {16, 16, 8, 8, 8, 8, 16, 16};
    const int gl[8] = {0, 0, 0, 0, 0, 0, 1, 1};
    const int gp[8] = {1, -1, 1, -1, 1, -1, -1, 1};
    int goff[8];
    { int o = 0; for (int i = 0; i < 8; i++) { goff[i] = o; o += gm[i] * (2 * gl[i] + 1); } }
    const int shoff[4] = {0, 1, 4, 9};

    int fan[8] = {0, 0, 0, 0, 0, 0, 0, 0};
    for (int i1 = 0; i1 < 4; i1++)
        for (int i2 = 0; i2 < 4; i2++)
            for (int i3 = 0; i3 < 8; i3++) {
                int l1 = i1, l2 = i2, l3 = gl[i3];
                int p1 = (l1 & 1) ? -1 : 1, p2 = (l2 & 1) ? -1 : 1;
                int dl = l1 - l2; if (dl < 0) dl = -dl;
                if (p1 * p2 == gp[i3] && dl <= l3 && l3 <= l1 + l2) fan[i3] += 1;
            }
    int pkbase[4][4][2];
    for (int a = 0; a < 4; a++) for (int b = 0; b < 4; b++) for (int c = 0; c < 2; c++) pkbase[a][b][c] = -1;

    float Ttmp[343];
    int pkcnt = 0, nT = 0, woff = 0;
    for (int i1 = 0; i1 < 4; i1++)
        for (int i2 = 0; i2 < 4; i2++)
            for (int i3 = 0; i3 < 8; i3++) {
                int l1 = i1, l2 = i2, l3 = gl[i3];
                int p1 = (l1 & 1) ? -1 : 1, p2 = (l2 & 1) ? -1 : 1;
                int dl = l1 - l2; if (dl < 0) dl = -dl;
                if (!(p1 * p2 == gp[i3] && dl <= l3 && l3 <= l1 + l2)) continue;
                if (pkbase[i1][i2][l3] < 0) {
                    pkbase[i1][i2][l3] = pkcnt;
                    pkcnt += 2 * l3 + 1;
                    real_cg_dev(l1, l2, l3, Ttmp);
                    int dd1 = 2 * l1 + 1, dd2 = 2 * l2 + 1, dd3 = 2 * l3 + 1;
                    for (int i = 0; i < dd1; i++)
                        for (int j = 0; j < dd2; j++)
                            for (int k = 0; k < dd3; k++) {
                                float cf = Ttmp[(i * dd2 + j) * dd3 + k];
                                if (fabsf(cf) > 1e-9f) {
                                    int4 en;
                                    en.x = shoff[l1] + i;
                                    en.y = shoff[l2] + j;
                                    en.z = pkbase[i1][i2][l3] + k;
                                    en.w = __float_as_int(cf);
                                    tprog[nT++] = en;
                                }
                            }
                }
                float alpha = sqrtf((float)(2 * l3 + 1) / (float)fan[i3]);
                int dd3 = 2 * l3 + 1;
                for (int w = 0; w < gm[i3]; w++) {
                    int4 we;
                    we.x = goff[i3] + w * dd3;
                    we.y = pkbase[i1][i2][l3];
                    we.z = dd3;
                    we.w = __float_as_int(alpha);
                    wprog[woff + w] = we;
                }
                woff += gm[i3];
            }
    // stable insertion sort by t-slot so K2 can accumulate in a register
    for (int i = 1; i < nT; i++) {
        int4 key = tprog[i];
        int j = i - 1;
        while (j >= 0 && tprog[j].z > key.z) { tprog[j + 1] = tprog[j]; j--; }
        tprog[j + 1] = key;
    }
    hdr->nT1 = nT;
    // TP2 constants: alpha = sqrt(1/32) (fan = 16+16)
    float a2 = sqrtf(1.0f / 32.0f);
    real_cg_dev(0, 0, 0, Ttmp);
    hdr->c000a = Ttmp[0] * a2;
    real_cg_dev(1, 1, 0, Ttmp);
    for (int i = 0; i < 3; i++)
        for (int j = 0; j < 3; j++)
            hdr->c110a[i * 3 + j] = Ttmp[i * 3 + j] * a2;
}

// =====================================================================
// K1 : per-edge geometry: sh(l<=3), d, scatter sh into x0
// =====================================================================
#define SQ3f  1.73205081f
#define SQ5f  2.23606798f
#define SQ7f  2.64575131f
#define SQ15f 3.87298335f
#define SQ42f 6.48074070f
#define SQ70f 8.36660027f
#define SQ105f 10.24695077f

__global__ void k1_edges(const float* __restrict__ pos, const int* __restrict__ esrc,
                         const int* __restrict__ edst, float* __restrict__ sh,
                         float* __restrict__ dbuf, float* __restrict__ x0, int E) {
    int e = blockIdx.x * blockDim.x + threadIdx.x;
    if (e >= E) return;
    int s = esrc[e], t = edst[e];
    float dx = pos[t * 3 + 0] - pos[s * 3 + 0];
    float dy = pos[t * 3 + 1] - pos[s * 3 + 1];
    float dz = pos[t * 3 + 2] - pos[s * 3 + 2];
    float d = sqrtf(dx * dx + dy * dy + dz * dz + 1e-12f);
    dbuf[e] = d;
    float inv = 1.0f / d;
    float x = dx * inv, y = dy * inv, z = dz * inv;
    float x2 = x * x, y2 = y * y, z2 = z * z;
    float v[16];
    v[0] = 1.0f;
    v[1] = SQ3f * y;  v[2] = SQ3f * z;  v[3] = SQ3f * x;
    v[4] = SQ15f * x * y; v[5] = SQ15f * y * z; v[6] = 0.5f * SQ5f * (3.0f * z2 - 1.0f);
    v[7] = SQ15f * x * z; v[8] = 0.5f * SQ15f * (x2 - y2);
    v[9]  = 0.25f * SQ70f * y * (3.0f * x2 - y2);
    v[10] = SQ105f * x * y * z;
    v[11] = 0.25f * SQ42f * y * (5.0f * z2 - 1.0f);
    v[12] = 0.5f * SQ7f * z * (5.0f * z2 - 3.0f);
    v[13] = 0.25f * SQ42f * x * (5.0f * z2 - 1.0f);
    v[14] = 0.5f * SQ105f * z * (x2 - y2);
    v[15] = 0.25f * SQ70f * x * (x2 - 3.0f * y2);
    v4f* shp = (v4f*)(sh + (size_t)e * 16);
    shp[0] = (v4f){v[0], v[1], v[2], v[3]};
    shp[1] = (v4f){v[4], v[5], v[6], v[7]};
    shp[2] = (v4f){v[8], v[9], v[10], v[11]};
    shp[3] = (v4f){v[12], v[13], v[14], v[15]};
    #pragma unroll
    for (int j = 0; j < 16; j++) atomAddF(&x0[(size_t)t * 16 + j], v[j]);
}

// =====================================================================
// A-operand builder: radial embed + layer-1 MLP, bf16 A-layout regs.
// k order used here (B uses the same): lanes 0-15 -> elems 0..15 = K 32s+0..15,
// lanes 16-31 -> elems 0..15 = K 32s+16..31.
// =====================================================================
__device__ __forceinline__ void build_A_regs(const float* __restrict__ w1,
                                             float d, int hl, v16bf A[8]) {
    float emb[10];
    const float step = 3.2f / 11.0f;
    const float istep = 11.0f / 3.2f;
    #pragma unroll
    for (int r = 0; r < 10; r++) {
        float u = (d - (float)(r + 1) * step) * istep;
        float u2 = u * u;
        float f = 0.0f;
        if (u2 < 1.0f) f = 1.14136f * __expf(-(u2 / fmaxf(1.0f - u2, 1e-6f)));
        emb[r] = f;   // sqrt(NUM_BASIS) * (1/sqrt(10)) cancels exactly
    }
    #pragma unroll
    for (int s = 0; s < 8; s++) {
        int b0 = 32 * s + hl * 16;
        v8f ac0 = {}, ac1 = {};
        #pragma unroll
        for (int r = 0; r < 10; r++) {
            const float* row = w1 + r * 256 + b0;
            v8f wlo = *(const v8f*)row;
            v8f whi = *(const v8f*)(row + 8);
            ac0 += wlo * emb[r];
            ac1 += whi * emb[r];
        }
        v16bf av;
        #pragma unroll
        for (int t = 0; t < 8; t++) {
            av[t]     = (__bf16)(fmaxf(ac0[t], 0.0f) * 1.41421356f);
            av[t + 8] = (__bf16)(fmaxf(ac1[t], 0.0f) * 1.41421356f);
        }
        A[s] = av;
    }
}

// =====================================================================
// K2 : fused  GEMM1 (WMMA bf16, LDS-staged B)  +  TP1  +  scatter
// block = 128 threads = 4 waves, wave = 16-edge tile
// LDS: t(8K) + ef(40K) + dst(256B) + union{a+b (8K) | Bstage (8K)}
// =====================================================================
__launch_bounds__(128)
__global__ void k2_tp1(const float* __restrict__ dbuf, const float* __restrict__ sh,
                       const int* __restrict__ esrc, const int* __restrict__ edst,
                       const float* __restrict__ w1,          // w_fc1_1 10x256
                       const __bf16* __restrict__ W2A,        // 272x256 bf16 N-major
                       const TPHdr* __restrict__ hdr,
                       const int4* __restrict__ tprog, const int4* __restrict__ wprog,
                       const float* __restrict__ x0, float* __restrict__ x1,
                       int E, float inv_nn) {
    __shared__ float t_ws[4][16][32];            // 8 KB, persistent
    __shared__ float ef_ws[4][16][DIM_X1];       // 40 KB, persistent
    __shared__ int   dst_ws[4][16];
    __shared__ __align__(16) char upool[8192];   // a/b (CG phase)  <->  B stage (GEMM)

    float (*a_ws)[16][16] = reinterpret_cast<float(*)[16][16]>(upool);          // [4][16][16]
    float (*b_ws)[16][16] = reinterpret_cast<float(*)[16][16]>(upool + 4096);
    __bf16 (*Bst)[256]    = reinterpret_cast<__bf16(*)[256]>(upool);            // [16][256]

    int lane = threadIdx.x & 31;
    int wv = threadIdx.x >> 5;
    int hl = lane >> 4;
    int lm = lane & 15;
    int eraw = (blockIdx.x * 4 + wv) * 16 + lm;
    int e = (eraw < E) ? eraw : 0;

    // zero per-wave edge-feature accumulator
    float* efflat = &ef_ws[wv][0][0];
    for (int i = lane; i < 16 * DIM_X1; i += 32) efflat[i] = 0.0f;

    if (hl == 0) {
        const v4f* shp = (const v4f*)(sh + (size_t)e * 16);
        *(v4f*)&b_ws[wv][lm][0]  = shp[0];
        *(v4f*)&b_ws[wv][lm][4]  = shp[1];
        *(v4f*)&b_ws[wv][lm][8]  = shp[2];
        *(v4f*)&b_ws[wv][lm][12] = shp[3];
        int src = esrc[e];
        const v4f* xp = (const v4f*)(x0 + (size_t)src * 16);
        *(v4f*)&a_ws[wv][lm][0]  = xp[0] * inv_nn;
        *(v4f*)&a_ws[wv][lm][4]  = xp[1] * inv_nn;
        *(v4f*)&a_ws[wv][lm][8]  = xp[2] * inv_nn;
        *(v4f*)&a_ws[wv][lm][12] = xp[3] * inv_nn;
        dst_ws[wv][lm] = (eraw < E) ? edst[e] : -1;
        #pragma unroll
        for (int k = 0; k < 32; k++) t_ws[wv][lm][k] = 0.0f;
    }

    // CG contraction program (sorted by slot): register accumulate, one store per slot
    int nT = hdr->nT1;
    if (hl == 0) {
        float acc = 0.0f;
        int cur = -1;
        for (int idx = 0; idx < nT; idx++) {
            int4 en = tprog[idx];
            if (en.z != cur) {
                if (cur >= 0) t_ws[wv][lm][cur] = acc;
                cur = en.z; acc = 0.0f;
            }
            acc += a_ws[wv][lm][en.x] * b_ws[wv][lm][en.y] * __int_as_float(en.w);
        }
        if (cur >= 0) t_ws[wv][lm][cur] = acc;
    }

    // layer-1 MLP -> bf16 A operand (registers)
    float d = dbuf[e];
    v16bf A[8];
    build_A_regs(w1, d, hl, A);

    // GEMM over 17 weight-column tiles; B tile staged in LDS per block
    for (int nt = 0; nt < 17; nt++) {
        __syncthreads();   // previous consumers done (and CG phase done for nt==0)
        {
            const uint4* gsrc = (const uint4*)(W2A + (size_t)nt * 16 * 256);
            uint4* ldst = (uint4*)upool;
            #pragma unroll
            for (int i = 0; i < 4; i++)
                ldst[threadIdx.x + 128 * i] = gsrc[threadIdx.x + 128 * i];
        }
        __syncthreads();   // stage complete

        // batch all 16 ds_loads, fence, then 8 back-to-back WMMAs
        const __bf16* brow = &Bst[lm][hl * 16];
        v16bf Bt[8];
        #pragma unroll
        for (int s = 0; s < 8; s++) Bt[s] = *(const v16bf*)(brow + 32 * s);
        SCHED_FENCE();

        v8f c = {};
        #pragma unroll
        for (int s = 0; s < 8; s++)
            c = __builtin_amdgcn_wmma_f32_16x16x32_bf16(false, A[s], false, Bt[s],
                                                        (short)0, c, false, false);
        SCHED_FENCE();

        int n = nt * 16 + lm;
        int4 wp = wprog[n];
        float alpha = __int_as_float(wp.w);
        #pragma unroll
        for (int r = 0; r < 8; r++) {
            int m = r + hl * 8;
            float cv = c[r] * alpha;
            for (int k = 0; k < wp.z; k++)
                ef_ws[wv][m][wp.x + k] += cv * t_ws[wv][m][wp.y + k];
        }
    }
    __syncthreads();

    // scatter-add into node features
    for (int i = lane; i < 16 * DIM_X1; i += 32) {
        int m = i / DIM_X1, col = i - DIM_X1 * m;
        int dn = dst_ws[wv][m];
        if (dn >= 0) atomAddF(&x1[(size_t)dn * DIM_X1 + col], ef_ws[wv][m][col] * inv_nn);
    }
}

// =====================================================================
// K3 : gate activation  (N x 160 -> N x 128)
// =====================================================================
__global__ void k3_gate(const float* __restrict__ x1, float* __restrict__ xg, int N) {
    int nd = blockIdx.x * blockDim.x + threadIdx.x;
    if (nd >= N) return;
    const float* x = x1 + (size_t)nd * DIM_X1;
    float* o = xg + (size_t)nd * DIM_XG;
    #pragma unroll
    for (int i = 0; i < 16; i++) o[i] = fmaxf(x[i], 0.0f);
    #pragma unroll
    for (int i = 16; i < 32; i++) o[i] = tanhf(x[i]);
    float g[32];
    #pragma unroll
    for (int j = 0; j < 8; j++) g[j]      = fmaxf(x[32 + j], 0.0f);
    #pragma unroll
    for (int j = 0; j < 8; j++) g[8 + j]  = tanhf(x[40 + j]);
    #pragma unroll
    for (int j = 0; j < 8; j++) g[16 + j] = fmaxf(x[48 + j], 0.0f);
    #pragma unroll
    for (int j = 0; j < 8; j++) g[24 + j] = tanhf(x[56 + j]);
    #pragma unroll
    for (int u = 0; u < 32; u++)
        for (int i = 0; i < 3; i++)
            o[32 + 3 * u + i] = x[64 + 3 * u + i] * g[u];
}

// =====================================================================
// K4 : fused  GEMM2 (WMMA bf16, whole B in LDS)  +  TP2  +  pooling
// =====================================================================
__launch_bounds__(128)
__global__ void k4_tp2(const float* __restrict__ dbuf, const float* __restrict__ sh,
                       const int* __restrict__ esrc, const int* __restrict__ edst,
                       const int* __restrict__ batch,
                       const float* __restrict__ w21,       // w_fc2_1 10x256
                       const __bf16* __restrict__ W2B,      // 32x256 bf16 N-major
                       const TPHdr* __restrict__ hdr,
                       const float* __restrict__ xg, float* __restrict__ out,
                       int E, float outscale) {
    __shared__ float s_ws[4][16][32];
    __shared__ float oacc[4][16];
    __shared__ int   bidx[4][16];
    __shared__ __align__(16) __bf16 Bst2[32][256];    // 16 KB, whole W2B

    int lane = threadIdx.x & 31;
    int wv = threadIdx.x >> 5;
    int hl = lane >> 4;
    int lm = lane & 15;
    int eraw = (blockIdx.x * 4 + wv) * 16 + lm;
    int e = (eraw < E) ? eraw : 0;

    // stage whole B once per block
    {
        const uint4* gsrc = (const uint4*)W2B;
        uint4* ldst = (uint4*)Bst2;
        for (int i = threadIdx.x; i < 1024; i += 128) ldst[i] = gsrc[i];
    }

    if (hl == 0) {
        int src = esrc[e];
        const float* xr = xg + (size_t)src * DIM_XG;
        float c000a = hdr->c000a;
        float c110[9];
        #pragma unroll
        for (int i = 0; i < 9; i++) c110[i] = hdr->c110a[i];
        float sh0 = sh[(size_t)e * 16 + 0];
        float b0 = sh[(size_t)e * 16 + 1];
        float b1 = sh[(size_t)e * 16 + 2];
        float b2 = sh[(size_t)e * 16 + 3];
        #pragma unroll
        for (int u = 0; u < 16; u++)
            s_ws[wv][lm][u] = xr[u] * sh0 * c000a;
        #pragma unroll
        for (int u = 0; u < 16; u++) {
            float ax = xr[32 + 3 * u], ay = xr[33 + 3 * u], az = xr[34 + 3 * u];
            s_ws[wv][lm][16 + u] =
                ax * (c110[0] * b0 + c110[1] * b1 + c110[2] * b2) +
                ay * (c110[3] * b0 + c110[4] * b1 + c110[5] * b2) +
                az * (c110[6] * b0 + c110[7] * b1 + c110[8] * b2);
        }
        oacc[wv][lm] = 0.0f;
        bidx[wv][lm] = (eraw < E) ? batch[edst[e]] : -1;
    }
    __syncthreads();

    float d = dbuf[e];
    v16bf A[8];
    build_A_regs(w21, d, hl, A);

    for (int nt = 0; nt < 2; nt++) {
        int n = nt * 16 + lm;
        const __bf16* brow = &Bst2[n][hl * 16];
        v16bf Bt[8];
        #pragma unroll
        for (int s = 0; s < 8; s++) Bt[s] = *(const v16bf*)(brow + 32 * s);
        SCHED_FENCE();

        v8f c = {};
        #pragma unroll
        for (int s = 0; s < 8; s++)
            c = __builtin_amdgcn_wmma_f32_16x16x32_bf16(false, A[s], false, Bt[s],
                                                        (short)0, c, false, false);
        SCHED_FENCE();

        #pragma unroll
        for (int r = 0; r < 8; r++) {
            int m = r + hl * 8;
            atomicAdd(&oacc[wv][m], c[r] * s_ws[wv][m][n]);
        }
    }
    __syncthreads();

    if (hl == 0 && bidx[wv][lm] >= 0)
        atomAddF(&out[bidx[wv][lm]], oacc[wv][lm] * outscale);
}

// =====================================================================
// host launcher
// =====================================================================
extern "C" void kernel_launch(void* const* d_in, const int* in_sizes, int n_in,
                              void* d_out, int out_size, void* d_ws, size_t ws_size,
                              hipStream_t stream) {
    const float* pos   = (const float*)d_in[0];
    const float* wfc11 = (const float*)d_in[1];
    const float* wfc12 = (const float*)d_in[2];
    const float* wfc21 = (const float*)d_in[3];
    const float* wfc22 = (const float*)d_in[4];
    const int*   esrc  = (const int*)d_in[5];
    const int*   edst  = (const int*)d_in[6];
    const int*   batch = (const int*)d_in[7];
    int N = in_sizes[0] / 3;
    int E = in_sizes[5];
    int G = out_size;
    float* out = (float*)d_out;

    char* ws = (char*)d_ws;
    size_t off = 0;
    auto take = [&](size_t b) { size_t o = off; off = (off + b + 255) & ~(size_t)255; return o; };
    size_t oHdr = take(sizeof(TPHdr));
    size_t oT   = take(1024 * 16);
    size_t oW   = take(TP1_NW * 16);
    size_t oA   = take((size_t)TP1_NW * 256 * 2);
    size_t oB   = take((size_t)TP2_NW * 256 * 2);
    size_t oSH  = take((size_t)E * 16 * 4);
    size_t oD   = take((size_t)E * 4);
    size_t oX0  = take((size_t)N * 16 * 4);
    size_t oX1  = take((size_t)N * DIM_X1 * 4);
    size_t oXG  = take((size_t)N * DIM_XG * 4);
    (void)ws_size; (void)n_in;

    TPHdr*  hdr   = (TPHdr*)(ws + oHdr);
    int4*   tprog = (int4*)(ws + oT);
    int4*   wprog = (int4*)(ws + oW);
    __bf16* W2A   = (__bf16*)(ws + oA);
    __bf16* W2B   = (__bf16*)(ws + oB);
    float*  shb   = (float*)(ws + oSH);
    float*  dbuf  = (float*)(ws + oD);
    float*  x0    = (float*)(ws + oX0);
    float*  x1    = (float*)(ws + oX1);
    float*  xgb   = (float*)(ws + oXG);

    float inv_nn  = 1.0f / sqrtf((float)E / (float)N);
    float invSqN  = 1.0f / sqrtf((float)N);
    float outscale = inv_nn * invSqN;

    int zt = N * DIM_X1;
    if (TP1_NW * 256 > zt) zt = TP1_NW * 256;
    k0_zero<<<(zt + 255) / 256, 256, 0, stream>>>(x0, x1, out, W2A, W2B, wfc12, wfc22, N, G);
    k0_prog<<<1, 1, 0, stream>>>(hdr, tprog, wprog);
    k1_edges<<<(E + 255) / 256, 256, 0, stream>>>(pos, esrc, edst, shb, dbuf, x0, E);
    k2_tp1<<<(E + 63) / 64, 128, 0, stream>>>(dbuf, shb, esrc, edst, wfc11, W2A, hdr,
                                              tprog, wprog, x0, x1, E, inv_nn);
    k3_gate<<<(N + 127) / 128, 128, 0, stream>>>(x1, xgb, N);
    k4_tp2<<<(E + 63) / 64, 128, 0, stream>>>(dbuf, shb, esrc, edst, batch, wfc21, W2B,
                                              hdr, xgb, out, E, outscale);
}
